// GPT_1425929142363
// MI455X (gfx1250) — compile-verified, matrix operand order checked by reading, not run on hardware
//
#include <hip/hip_runtime.h>
#include <hip/hip_bf16.h>
#include <math.h>

#define NL 8
#define NH 16
#define C 1024
#define C3 3072
#define TT 1025
#define BB 4
#define MM (BB * TT)   // 4100 rows (tokens)
#define VP 32001
#define HD 64
#define LDP 72         // LDS row stride in ushorts: 144 B = 36 banks (16B-aligned)

typedef __attribute__((ext_vector_type(16))) __bf16          v16bf;
typedef __attribute__((ext_vector_type(8)))  __bf16          v8bf;
typedef __attribute__((ext_vector_type(16))) unsigned short  v16u;
typedef __attribute__((ext_vector_type(8)))  unsigned short  v8u;
typedef __attribute__((ext_vector_type(8)))  float           v8f;

// Hardware RNE f32 -> bf16 (clang lowers the cast to v_cvt on gfx1250)
static __device__ __forceinline__ unsigned short f2bf(float f) {
  __bf16 b = (__bf16)f;
  return __builtin_bit_cast(unsigned short, b);
}

static __device__ __forceinline__ v8u pack8(float4 a, float4 b) {
  v8bf r;
  r[0] = (__bf16)a.x; r[1] = (__bf16)a.y; r[2] = (__bf16)a.z; r[3] = (__bf16)a.w;
  r[4] = (__bf16)b.x; r[5] = (__bf16)b.y; r[6] = (__bf16)b.z; r[7] = (__bf16)b.w;
  return __builtin_bit_cast(v8u, r);
}

// ---------------------------------------------------------------- embedding
__global__ void embed_kernel(const int* __restrict__ idx, const float* __restrict__ wte,
                             const float* __restrict__ wpe, float* __restrict__ x) {
  size_t i = (size_t)blockIdx.x * blockDim.x + threadIdx.x;       // float4 index
  if (i >= (size_t)MM * (C / 4)) return;
  int c4  = (int)(i % (C / 4));
  int row = (int)(i / (C / 4));
  int t   = row % TT;
  const float4* wte4 = (const float4*)wte;
  const float4* wpe4 = (const float4*)wpe;
  float4 a = wte4[(size_t)idx[row] * (C / 4) + c4];
  float4 b = wpe4[(size_t)t * (C / 4) + c4];
  float4 o; o.x = a.x + b.x; o.y = a.y + b.y; o.z = a.z + b.z; o.w = a.w + b.w;
  ((float4*)x)[i] = o;
}

// ---------------------------------------------------------------- layernorm (one wave32 per row)
__global__ void ln_kernel(const float* __restrict__ in, const float* __restrict__ w,
                          const float* __restrict__ b, float* __restrict__ out, int rows) {
  int wave = threadIdx.x >> 5, lane = threadIdx.x & 31;
  int row = blockIdx.x * 8 + wave;
  if (row >= rows) return;
  const float4* r4 = (const float4*)(in + (size_t)row * C);
  float s = 0.f, sq = 0.f;
  float4 v[8];
  #pragma unroll
  for (int j = 0; j < 8; ++j) {
    v[j] = r4[lane + j * 32];
    s  += v[j].x + v[j].y + v[j].z + v[j].w;
    sq += v[j].x * v[j].x + v[j].y * v[j].y + v[j].z * v[j].z + v[j].w * v[j].w;
  }
  #pragma unroll
  for (int m = 16; m >= 1; m >>= 1) { s += __shfl_xor(s, m, 32); sq += __shfl_xor(sq, m, 32); }
  float mu   = s * (1.f / C);
  float var  = sq * (1.f / C) - mu * mu;
  float rstd = rsqrtf(var + 1e-5f);
  const float4* w4 = (const float4*)w;
  const float4* b4 = (const float4*)b;
  float4* o4 = (float4*)(out + (size_t)row * C);
  #pragma unroll
  for (int j = 0; j < 8; ++j) {
    int c = lane + j * 32;
    float4 ww = w4[c], bb = b4[c], o;
    o.x = (v[j].x - mu) * rstd * ww.x + bb.x;
    o.y = (v[j].y - mu) * rstd * ww.y + bb.y;
    o.z = (v[j].z - mu) * rstd * ww.z + bb.z;
    o.w = (v[j].w - mu) * rstd * ww.w + bb.w;
    o4[c] = o;
  }
}

// ---------------------------------------------------------------- bf16 WMMA GEMM
// Cout[M,N] = act( A[M,K] @ W + bias ) (+ resid)
// W is [K,N] (requires N % 128 == 0) or wTrans [N,K] (any N; rows clamped).
// Requires K % 64 == 0. Block tile 64(M) x 128(N), K-step 64, 8 waves,
// wave tile 32x32 -> 8 WMMAs per LDS fill, edge handling via index clamping.
__global__ __launch_bounds__(256) void gemm_wmma(
    const float* __restrict__ A, const float* __restrict__ W,
    const float* __restrict__ bias, const float* __restrict__ resid,
    float* __restrict__ Cout, int Mrows, int Ncols, int Kdim,
    int wTrans, int doGelu) {
  __shared__ unsigned short As[64][LDP];    // bf16 bits, [m][k] — K-contiguous
  __shared__ unsigned short Bs[128][LDP];   // bf16 bits, [n][k] — K-contiguous
  const int tid  = threadIdx.x;
  const int lane = tid & 31, wave = tid >> 5;
  const int wm2 = wave & 1;        // 2 waves along M  (2 x 32 = 64)
  const int wn4 = wave >> 1;       // 4 waves along N  (4 x 32 = 128)
  const int hi16 = (lane >> 4) & 1;
  const int l16  = lane & 15;
  const int m0 = blockIdx.y * 64, n0 = blockIdx.x * 128;

  v8f acc[2][2] = {};

  // Fragment: two ds_load_b128 from a K-contiguous bf16 row + shuffle.
  auto frag = [&](const unsigned short* p) -> v16bf {
    v8u lo = *(const v8u*)(p + hi16 * 8);
    v8u hi = *(const v8u*)(p + 16 + hi16 * 8);
    v16u cc = __builtin_shufflevector(lo, hi, 0, 1, 2, 3, 4, 5, 6, 7,
                                               8, 9, 10, 11, 12, 13, 14, 15);
    return __builtin_bit_cast(v16bf, cc);
  };

  auto computeKs = [&](int ks) {   // ks = 0,1 selects 32-wide K sub-step
    const int ko = ks * 32;
    v16bf aF0 = frag(&As[wm2 * 32 + l16][ko]);
    v16bf aF1 = frag(&As[wm2 * 32 + 16 + l16][ko]);
    v16bf bF0 = frag(&Bs[wn4 * 32 + l16][ko]);
    v16bf bF1 = frag(&Bs[wn4 * 32 + 16 + l16][ko]);
    acc[0][0] = __builtin_amdgcn_wmma_f32_16x16x32_bf16(false, aF0, false, bF0, (short)0, acc[0][0], false, false);
    acc[0][1] = __builtin_amdgcn_wmma_f32_16x16x32_bf16(false, aF0, false, bF1, (short)0, acc[0][1], false, false);
    acc[1][0] = __builtin_amdgcn_wmma_f32_16x16x32_bf16(false, aF1, false, bF0, (short)0, acc[1][0], false, false);
    acc[1][1] = __builtin_amdgcn_wmma_f32_16x16x32_bf16(false, aF1, false, bF1, (short)0, acc[1][1], false, false);
  };

  // Per-thread fill coordinates (row indices clamped; OOB rows feed only
  // accumulator slots that the guarded epilogue never stores).
  int mA = m0 + (tid >> 2);                 // A: 64 rows x 64 k
  if (mA >= Mrows) mA = Mrows - 1;
  const int kA = (tid & 3) * 16;
  const int kB0 = (tid >> 3) * 2, nB0 = (tid & 7) * 16;   // B normal: 64 k x 128 n
  int nT = n0 + (tid >> 1);                               // B trans : 128 n x 64 k
  if (nT >= Ncols) nT = Ncols - 1;
  const int kT = (tid & 1) * 32;

  float4 ar[4], br[8];

  auto loadTile = [&](int kk) {
    { const float4* p = (const float4*)&A[(size_t)mA * Kdim + kk + kA];
      ar[0] = p[0]; ar[1] = p[1]; ar[2] = p[2]; ar[3] = p[3]; }
    if (wTrans) {
      const float4* p = (const float4*)&W[(size_t)nT * Kdim + kk + kT];
      #pragma unroll
      for (int i = 0; i < 8; ++i) br[i] = p[i];
    } else {
      const float4* p0 = (const float4*)&W[(size_t)(kk + kB0) * Ncols + n0 + nB0];
      const float4* p1 = (const float4*)&W[(size_t)(kk + kB0 + 1) * Ncols + n0 + nB0];
      #pragma unroll
      for (int i = 0; i < 4; ++i) { br[i] = p0[i]; br[4 + i] = p1[i]; }
    }
  };

  auto storeLDS = [&]() {
    *(v8u*)&As[tid >> 2][kA]     = pack8(ar[0], ar[1]);
    *(v8u*)&As[tid >> 2][kA + 8] = pack8(ar[2], ar[3]);
    if (wTrans) {
      unsigned short* q = &Bs[tid >> 1][kT];
      *(v8u*)(q)      = pack8(br[0], br[1]);
      *(v8u*)(q + 8)  = pack8(br[2], br[3]);
      *(v8u*)(q + 16) = pack8(br[4], br[5]);
      *(v8u*)(q + 24) = pack8(br[6], br[7]);
    } else {
      const float* f = (const float*)br;
      #pragma unroll
      for (int i = 0; i < 16; ++i) Bs[nB0 + i][kB0]     = f2bf(f[i]);
      #pragma unroll
      for (int i = 0; i < 16; ++i) Bs[nB0 + i][kB0 + 1] = f2bf(f[16 + i]);
    }
  };

  loadTile(0);
  for (int kk = 0; kk < Kdim; kk += 64) {
    storeLDS();
    __syncthreads();
    if (kk + 64 < Kdim) loadTile(kk + 64);   // overlap next global burst w/ WMMAs
    computeKs(0);
    computeKs(1);
    __syncthreads();
  }

  // -------- epilogue: C layout — VGPR r: M = r (+8 for hi lanes); N = lane&15.
  #pragma unroll
  for (int s = 0; s < 2; ++s) {
    #pragma unroll
    for (int r = 0; r < 8; ++r) {
      int m = m0 + wm2 * 32 + s * 16 + r + hi16 * 8;
      if (m >= Mrows) continue;
      #pragma unroll
      for (int t = 0; t < 2; ++t) {
        int n = n0 + wn4 * 32 + t * 16 + l16;
        if (n >= Ncols) continue;
        float v = acc[s][t][r];
        if (bias) v += bias[n];
        if (doGelu) {
          float xv = v;
          float inner = 0.7978845608028654f * (xv + 0.044715f * xv * xv * xv);
          v = 0.5f * xv * (1.f + tanhf(inner));
        }
        if (resid) v += resid[(size_t)m * Ncols + n];
        Cout[(size_t)m * Ncols + n] = v;
      }
    }
  }
}

// ---------------------------------------------------------------- attention (online softmax, 1 query/lane)
__global__ void attn_kernel(const float* __restrict__ qkv, float* __restrict__ out) {
  int bh = blockIdx.x;
  int b = bh / NH, hh = bh % NH;
  int q = blockIdx.y * blockDim.x + threadIdx.x;
  if (q >= TT) return;
  const float4* qp = (const float4*)(qkv + (size_t)(b * TT + q) * C3 + hh * HD);
  float4 qr[16], acc[16];
  #pragma unroll
  for (int d = 0; d < 16; ++d) { qr[d] = qp[d]; acc[d] = make_float4(0.f, 0.f, 0.f, 0.f); }
  float mx = -INFINITY, sum = 0.f;
  const float scale = 0.0125f;   // 0.1/sqrt(64)
  for (int k = 0; k <= q; ++k) {
    const float4* kp = (const float4*)(qkv + (size_t)(b * TT + k) * C3 + C + hh * HD);
    float s = 0.f;
    #pragma unroll
    for (int d = 0; d < 16; ++d) {
      float4 kv = kp[d];
      s += qr[d].x * kv.x + qr[d].y * kv.y + qr[d].z * kv.z + qr[d].w * kv.w;
    }
    s *= scale;
    float mnew = fmaxf(mx, s);
    float corr = __expf(mx - mnew);
    float p    = __expf(s - mnew);
    const float4* vp = (const float4*)(qkv + (size_t)(b * TT + k) * C3 + 2 * C + hh * HD);
    #pragma unroll
    for (int d = 0; d < 16; ++d) {
      float4 vv = vp[d];
      acc[d].x = acc[d].x * corr + p * vv.x;
      acc[d].y = acc[d].y * corr + p * vv.y;
      acc[d].z = acc[d].z * corr + p * vv.z;
      acc[d].w = acc[d].w * corr + p * vv.w;
    }
    sum = sum * corr + p;
    mx = mnew;
  }
  float inv = 1.f / sum;
  float4* op = (float4*)(out + (size_t)(b * TT + q) * C + hh * HD);
  #pragma unroll
  for (int d = 0; d < 16; ++d) {
    float4 o;
    o.x = acc[d].x * inv; o.y = acc[d].y * inv; o.z = acc[d].z * inv; o.w = acc[d].w * inv;
    op[d] = o;
  }
}

// ---------------------------------------------------------------- loss
__global__ void zero_kernel(float* p) {
  if (threadIdx.x == 0 && blockIdx.x == 0) p[0] = 0.f;
}

__global__ void loss_kernel(const float* __restrict__ logits, const int* __restrict__ idx,
                            float* __restrict__ lossbuf) {
  __shared__ float red[256];
  int r = blockIdx.x;                 // 0..2047
  int b = r >> 9, t = 512 + (r & 511);
  const float* row = logits + (size_t)(b * TT + t) * VP;
  const float4* row4 = (const float4*)row;
  int tid = threadIdx.x;

  float mx = -INFINITY;
  for (int c = tid; c < VP / 4; c += 256) {
    float4 v = row4[c];
    mx = fmaxf(mx, fmaxf(fmaxf(v.x, v.y), fmaxf(v.z, v.w)));
  }
  if (tid == 0) mx = fmaxf(mx, row[VP - 1]);
  red[tid] = mx; __syncthreads();
  for (int s = 128; s >= 1; s >>= 1) {
    if (tid < s) red[tid] = fmaxf(red[tid], red[tid + s]);
    __syncthreads();
  }
  mx = red[0]; __syncthreads();

  float sum = 0.f;
  for (int c = tid; c < VP / 4; c += 256) {
    float4 v = row4[c];
    sum += __expf(v.x - mx) + __expf(v.y - mx) + __expf(v.z - mx) + __expf(v.w - mx);
  }
  if (tid == 0) sum += __expf(row[VP - 1] - mx);
  red[tid] = sum; __syncthreads();
  for (int s = 128; s >= 1; s >>= 1) {
    if (tid < s) red[tid] += red[tid + s];
    __syncthreads();
  }
  if (tid == 0) {
    int tgt = idx[b * TT + t + 1];
    float lp = row[tgt] - mx - __logf(red[0]);
    atomicAdd(lossbuf, lp);
  }
}

__global__ void finalize_kernel(const float* lossbuf, float* dloss) {
  if (threadIdx.x == 0) dloss[0] = -lossbuf[0] / 2048.f;
}

// ---------------------------------------------------------------- driver
extern "C" void kernel_launch(void* const* d_in, const int* in_sizes, int n_in,
                              void* d_out, int out_size, void* d_ws, size_t ws_size,
                              hipStream_t stream) {
  (void)in_sizes; (void)n_in; (void)out_size; (void)ws_size;
  const int*   idx   = (const int*)d_in[0];
  const float* wte   = (const float*)d_in[1];
  const float* wpe   = (const float*)d_in[2];
  const float* ln1w  = (const float*)d_in[3];
  const float* ln1b  = (const float*)d_in[4];
  const float* attnw = (const float*)d_in[5];
  const float* attnb = (const float*)d_in[6];
  const float* projw = (const float*)d_in[7];
  const float* projb = (const float*)d_in[8];
  const float* ln2w  = (const float*)d_in[9];
  const float* ln2b  = (const float*)d_in[10];
  const float* fc1w  = (const float*)d_in[11];
  const float* fc1b  = (const float*)d_in[12];
  const float* fc2w  = (const float*)d_in[13];
  const float* fc2b  = (const float*)d_in[14];
  const float* lnfw  = (const float*)d_in[15];
  const float* lnfb  = (const float*)d_in[16];

  float* logits = (float*)d_out;
  size_t MC = (size_t)MM * C;
  float* x    = (float*)d_ws;
  float* h    = x + MC;
  float* big  = h + MC;                       // [MM, 3072]
  float* lossbuf = big + (size_t)MM * C3;

  dim3 blk(256);
  embed_kernel<<<dim3((unsigned)((MM * (C / 4) + 255) / 256)), blk, 0, stream>>>(idx, wte, wpe, x);

  dim3 lnGrid((MM + 7) / 8);
  auto gg = [](int N, int M) { return dim3((N + 127) / 128, (M + 63) / 64); };

  for (int l = 0; l < NL; ++l) {
    ln_kernel<<<lnGrid, blk, 0, stream>>>(x, ln1w + l * C, ln1b + l * C, h, MM);
    gemm_wmma<<<gg(C3, MM), blk, 0, stream>>>(h, attnw + (size_t)l * C * C3,
                                              attnb + (size_t)l * C3, nullptr,
                                              big, MM, C3, C, 0, 0);
    attn_kernel<<<dim3(BB * NH, (TT + 255) / 256), blk, 0, stream>>>(big, h);
    gemm_wmma<<<gg(C, MM), blk, 0, stream>>>(h, projw + (size_t)l * C * C,
                                             projb + (size_t)l * C, x,
                                             x, MM, C, C, 0, 0);
    ln_kernel<<<lnGrid, blk, 0, stream>>>(x, ln2w + l * C, ln2b + l * C, h, MM);
    gemm_wmma<<<gg(C3, MM), blk, 0, stream>>>(h, fc1w + (size_t)l * C * C3,
                                              fc1b + (size_t)l * C3, nullptr,
                                              big, MM, C3, C, 0, 1);
    gemm_wmma<<<gg(C, MM), blk, 0, stream>>>(big, fc2w + (size_t)l * C3 * C,
                                             fc2b + (size_t)l * C, x,
                                             x, MM, C, C3, 0, 0);
  }

  ln_kernel<<<lnGrid, blk, 0, stream>>>(x, lnfw, lnfb, h, MM);
  // tied lm_head: logits = h @ wte^T  (W stored [V+1, C] -> wTrans)
  gemm_wmma<<<gg(VP, MM), blk, 0, stream>>>(h, wte, nullptr, nullptr,
                                            logits, MM, VP, C, 1, 0);

  zero_kernel<<<1, 32, 0, stream>>>(lossbuf);
  loss_kernel<<<dim3(2048), blk, 0, stream>>>(logits, idx, lossbuf);
  finalize_kernel<<<1, 32, 0, stream>>>(lossbuf, logits + (size_t)MM * VP);
}
